// myodefunc_31044023616347
// MI455X (gfx1250) — matrix operand order repaired and checked by reading, compiled, not run on hardware
//
#include <hip/hip_runtime.h>

// CDNA5 / gfx1250, wave32.
// Six 64x64 (K=64) GEMMs per 32-row wave strip via v_wmma_f32_16x16x32_f16,
// two 16-row M-strips per wave sharing each B fragment (2x B reuse).
//
// Fragment layout assumptions (CDNA5 ISA §7.12.2):
//   A (16x32 f16): lane = 16*kh + m (m=0..15). half e: g=e/8, w=e%8,
//                  K = g*16 + kh*8 + w.
//   B (32x16 f16): lane = 16*kh + n (n=0..15). half e: g=e/8, w=e%8, K = g*16 + kh*8 + w.
//   C/D (16x16 f32): lane L, vgpr v: m = v + 8*(L/16), n = L%16.

typedef __attribute__((ext_vector_type(16))) _Float16 v16h;
typedef __attribute__((ext_vector_type(8)))  _Float16 v8h;
typedef __attribute__((ext_vector_type(8)))  float    v8f;

#define NBLOCKS         512
#define WAVES_PER_BLOCK 8
#define ROWS_PER_WAVE   32                                   // two 16-row strips
#define ROWS_PER_TILE   (WAVES_PER_BLOCK * ROWS_PER_WAVE)    // 256

__device__ __forceinline__ v16h load_a_frag_f32(const float* __restrict__ rb,
                                                int kf, int hi) {
  // rb = row base (row = lane%16), stride-64 fp32 source (global z / e)
  v16h a;
#pragma unroll
  for (int g = 0; g < 2; ++g) {
#pragma unroll
    for (int w = 0; w < 8; w += 2) {
      float2 f = *(const float2*)(rb + kf * 32 + g * 16 + hi * 8 + w);
      a[g * 8 + w]     = (_Float16)f.x;
      a[g * 8 + w + 1] = (_Float16)f.y;
    }
  }
  return a;
}

__device__ __forceinline__ v16h load_a_frag_h(const _Float16* rb, int kf, int hi) {
  // rb = row base into per-wave f16 LDS bounce buffer (stride 64)
  v8h a0 = *(const v8h*)(rb + kf * 32 + hi * 8);        // K = kf*32 + kh*8 + 0..7
  v8h a1 = *(const v8h*)(rb + kf * 32 + 16 + hi * 8);   // K = kf*32 + 16 + kh*8 + 0..7
  return __builtin_shufflevector(a0, a1, 0, 1, 2, 3, 4, 5, 6, 7,
                                 8, 9, 10, 11, 12, 13, 14, 15);
}

__global__ __launch_bounds__(256) void flowdiv_wmma_kernel(
    const float* __restrict__ t,
    const float* __restrict__ z,
    const float* __restrict__ e,
    const float* __restrict__ W0, const float* __restrict__ b0,
    const float* __restrict__ W1, const float* __restrict__ b1,
    const float* __restrict__ W2, const float* __restrict__ b2,
    float* __restrict__ out_zdot,   // [B,64]
    float* __restrict__ out_div,    // [B]
    int Btotal)
{
  // 48 B-fragment slots: slot = gemm*8 + kf*4 + nt
  //   gemm 0..2: forward layer l   (B[k][n] = W_l[n][k+1])
  //   gemm 3..5: backward layer 2,1,0 (B[k][n] = W_l[k][n+1])
  __shared__ __align__(32) _Float16 sB[48][32][16];                 // 48 KB
  __shared__ float sBias[3][64];                                    // t*W[:,0]+b
  __shared__ __align__(16) _Float16 sScr[WAVES_PER_BLOCK][32][64];  // 32 KB bounce

  const int tid  = threadIdx.x;
  const int lane = tid & 31;
  const int wave = tid >> 5;
  const int hi   = lane >> 4;
  const int lm   = lane & 15;

  // ---- build WMMA B-fragment images of all six GEMM operands (once/block) --
  for (int idx = tid; idx < 48 * 512; idx += 256) {
    int slot = idx >> 9;
    int rem  = idx & 511;
    int ln = rem >> 4, h = rem & 15;
    int gemm = slot >> 3, kf = (slot >> 2) & 1, nt = slot & 3;
    int n = nt * 16 + (ln & 15);
    int K = kf * 32 + (h >> 3) * 16 + (ln >> 4) * 8 + (h & 7);
    int wl = (gemm < 3) ? gemm : (5 - gemm);
    const float* W = (wl == 0) ? W0 : (wl == 1) ? W1 : W2;
    float v = (gemm < 3) ? W[n * 65 + K + 1] : W[K * 65 + n + 1];
    sB[slot][ln][h] = (_Float16)v;
  }
  if (tid < 192) {
    int l = tid >> 6, n = tid & 63;
    const float* W  = (l == 0) ? W0 : (l == 1) ? W1 : W2;
    const float* bb = (l == 0) ? b0 : (l == 1) ? b1 : b2;
    sBias[l][n] = t[0] * W[n * 65] + bb[n];
  }
  __syncthreads();

  _Float16* scrh = &sScr[wave][0][0];
  const int ntiles = Btotal / ROWS_PER_TILE;

  for (int tile = blockIdx.x; tile < ntiles; tile += gridDim.x) {
    const size_t r0 = (size_t)tile * ROWS_PER_TILE + (size_t)wave * ROWS_PER_WAVE;

    // prefetch next grid-stride tile's z/e strips (global_prefetch_b8)
    {
      size_t rn = r0 + (size_t)gridDim.x * ROWS_PER_TILE;
      if (rn + ROWS_PER_WAVE <= (size_t)Btotal) {
        // 32 rows * 256 B = 8 KB per array -> 2 prefetches of 32 lanes*128 B
        __builtin_prefetch(z + rn * 64 + (size_t)lane * 32, 0, 1);
        __builtin_prefetch(z + rn * 64 + 4096 + (size_t)lane * 32, 0, 1);
        __builtin_prefetch(e + rn * 64 + (size_t)lane * 32, 0, 1);
        __builtin_prefetch(e + rn * 64 + 4096 + (size_t)lane * 32, 0, 1);
      }
    }

    v16h A[2][2];  // [mh][kf]
#pragma unroll
    for (int mh = 0; mh < 2; ++mh) {
      const float* zrow = z + (r0 + (size_t)(mh * 16 + lm)) * 64;
      A[mh][0] = load_a_frag_f32(zrow, 0, hi);
      A[mh][1] = load_a_frag_f32(zrow, 1, hi);
    }

    unsigned msk[2][2];  // [layer][mh]

    // -------------- forward layers 0,1 (bias + ReLU, record masks) ---------
#pragma unroll
    for (int l = 0; l < 2; ++l) {
      unsigned m0 = 0, m1 = 0;
#pragma unroll
      for (int nt = 0; nt < 4; ++nt) {
        v16h bf0 = *(const v16h*)&sB[l * 8 + nt][lane][0];
        v16h bf1 = *(const v16h*)&sB[l * 8 + 4 + nt][lane][0];
        float c0 = sBias[l][nt * 16 + lm];
#pragma unroll
        for (int mh = 0; mh < 2; ++mh) {
          v8f acc = {c0, c0, c0, c0, c0, c0, c0, c0};
          acc = __builtin_amdgcn_wmma_f32_16x16x32_f16(
              false, A[mh][0], false, bf0, (short)0, acc, false, false);
          acc = __builtin_amdgcn_wmma_f32_16x16x32_f16(
              false, A[mh][1], false, bf1, (short)0, acc, false, false);
          unsigned mm = 0;
#pragma unroll
          for (int v = 0; v < 8; ++v) {
            float p = acc[v];
            bool pos = p > 0.0f;
            mm |= (pos ? 1u : 0u) << (nt * 8 + v);
            scrh[(mh * 16 + v + 8 * hi) * 64 + nt * 16 + lm] =
                (_Float16)(pos ? p : 0.0f);
          }
          if (mh == 0) m0 |= mm; else m1 |= mm;
        }
      }
      msk[l][0] = m0; msk[l][1] = m1;
#pragma unroll
      for (int mh = 0; mh < 2; ++mh) {
        A[mh][0] = load_a_frag_h(scrh + (mh * 16 + lm) * 64, 0, hi);
        A[mh][1] = load_a_frag_h(scrh + (mh * 16 + lm) * 64, 1, hi);
      }
    }

    // -------------- forward layer 2 -> z_dot (fp32, C-layout stores) -------
#pragma unroll
    for (int nt = 0; nt < 4; ++nt) {
      v16h bf0 = *(const v16h*)&sB[16 + nt][lane][0];
      v16h bf1 = *(const v16h*)&sB[20 + nt][lane][0];
      float c0 = sBias[2][nt * 16 + lm];
#pragma unroll
      for (int mh = 0; mh < 2; ++mh) {
        v8f acc = {c0, c0, c0, c0, c0, c0, c0, c0};
        acc = __builtin_amdgcn_wmma_f32_16x16x32_f16(
            false, A[mh][0], false, bf0, (short)0, acc, false, false);
        acc = __builtin_amdgcn_wmma_f32_16x16x32_f16(
            false, A[mh][1], false, bf1, (short)0, acc, false, false);
#pragma unroll
        for (int v = 0; v < 8; ++v)
          out_zdot[(r0 + (size_t)(mh * 16 + v + 8 * hi)) * 64 + nt * 16 + lm] =
              acc[v];
      }
    }

    // -------------- backward: g = e; through layers 2,1 with masks ---------
#pragma unroll
    for (int mh = 0; mh < 2; ++mh) {
      const float* erow = e + (r0 + (size_t)(mh * 16 + lm)) * 64;
      A[mh][0] = load_a_frag_f32(erow, 0, hi);
      A[mh][1] = load_a_frag_f32(erow, 1, hi);
    }

#pragma unroll
    for (int p = 0; p < 2; ++p) {   // p=0: @W2b then mask1; p=1: @W1b then mask0
#pragma unroll
      for (int nt = 0; nt < 4; ++nt) {
        v16h bf0 = *(const v16h*)&sB[(3 + p) * 8 + nt][lane][0];
        v16h bf1 = *(const v16h*)&sB[(3 + p) * 8 + 4 + nt][lane][0];
#pragma unroll
        for (int mh = 0; mh < 2; ++mh) {
          v8f acc = {0.f, 0.f, 0.f, 0.f, 0.f, 0.f, 0.f, 0.f};
          acc = __builtin_amdgcn_wmma_f32_16x16x32_f16(
              false, A[mh][0], false, bf0, (short)0, acc, false, false);
          acc = __builtin_amdgcn_wmma_f32_16x16x32_f16(
              false, A[mh][1], false, bf1, (short)0, acc, false, false);
          unsigned mk = msk[1 - p][mh];
#pragma unroll
          for (int v = 0; v < 8; ++v) {
            float gv = ((mk >> (nt * 8 + v)) & 1u) ? acc[v] : 0.0f;
            scrh[(mh * 16 + v + 8 * hi) * 64 + nt * 16 + lm] = (_Float16)gv;
          }
        }
      }
#pragma unroll
      for (int mh = 0; mh < 2; ++mh) {
        A[mh][0] = load_a_frag_h(scrh + (mh * 16 + lm) * 64, 0, hi);
        A[mh][1] = load_a_frag_h(scrh + (mh * 16 + lm) * 64, 1, hi);
      }
    }

    // -------------- backward through layer 0 -> g_z, then divergence -------
#pragma unroll
    for (int mh = 0; mh < 2; ++mh) {
      v8f gz[4];
#pragma unroll
      for (int nt = 0; nt < 4; ++nt) {
        v16h bf0 = *(const v16h*)&sB[40 + nt][lane][0];
        v16h bf1 = *(const v16h*)&sB[44 + nt][lane][0];
        v8f acc = {0.f, 0.f, 0.f, 0.f, 0.f, 0.f, 0.f, 0.f};
        acc = __builtin_amdgcn_wmma_f32_16x16x32_f16(
            false, A[mh][0], false, bf0, (short)0, acc, false, false);
        acc = __builtin_amdgcn_wmma_f32_16x16x32_f16(
            false, A[mh][1], false, bf1, (short)0, acc, false, false);
        gz[nt] = acc;
      }
      // div[m] = sum_n gz[m,n]*e[m,n]; row m = mh*16 + v + 8*hi on lane half
#pragma unroll
      for (int v = 0; v < 8; ++v) {
        const float* er = e + (r0 + (size_t)(mh * 16 + v + 8 * hi)) * 64 + lm;
        float s = 0.0f;
#pragma unroll
        for (int nt = 0; nt < 4; ++nt)
          s += gz[nt][v] * er[nt * 16];
        s += __shfl_xor(s, 1, 32);
        s += __shfl_xor(s, 2, 32);
        s += __shfl_xor(s, 4, 32);
        s += __shfl_xor(s, 8, 32);
        if (lm == 0)
          out_div[r0 + (size_t)(mh * 16 + v + 8 * hi)] = -s;
      }
    }
  }
}

extern "C" void kernel_launch(void* const* d_in, const int* in_sizes, int n_in,
                              void* d_out, int out_size, void* d_ws, size_t ws_size,
                              hipStream_t stream) {
  const float* t  = (const float*)d_in[0];
  const float* z  = (const float*)d_in[1];
  const float* e  = (const float*)d_in[2];
  const float* W0 = (const float*)d_in[3];
  const float* b0 = (const float*)d_in[4];
  const float* W1 = (const float*)d_in[5];
  const float* b1 = (const float*)d_in[6];
  const float* W2 = (const float*)d_in[7];
  const float* b2 = (const float*)d_in[8];

  const int B = in_sizes[1] / 64;               // z is [B,64]
  float* out_zdot = (float*)d_out;              // [B,64]
  float* out_div  = out_zdot + (size_t)B * 64;  // [B]

  int ntiles = B / ROWS_PER_TILE;               // 262144/256 = 1024
  int blocks = (ntiles < NBLOCKS) ? ntiles : NBLOCKS;
  flowdiv_wmma_kernel<<<blocks, 256, 0, stream>>>(
      t, z, e, W0, b0, W1, b1, W2, b2, out_zdot, out_div, B);
}
